// SwiGLUExpert_81862076662210
// MI455X (gfx1250) — matrix operand order repaired and checked by reading, not compiled, not run on hardware
//
#include <hip/hip_runtime.h>
#include <stdint.h>

#define TOK_K 2048          // D_MODEL
#define EXP_N 8192          // D_EXPERT
#define QEPS  1e-8f

typedef __attribute__((ext_vector_type(16))) __bf16   bf16x16;
typedef __attribute__((ext_vector_type(8)))  __bf16   bf16x8;
typedef __attribute__((ext_vector_type(8)))  float    f32x8;
typedef __attribute__((ext_vector_type(4)))  uint32_t u32x4;
typedef __attribute__((ext_vector_type(8)))  int      i32x8;
typedef __attribute__((ext_vector_type(4)))  int      i32x4;

union BF16Frag { bf16x16 v; bf16x8 h[2]; };

__device__ __forceinline__ f32x8 wmma_bf16f32(bf16x16 a, bf16x16 b, f32x8 c) {
  // v_wmma_f32_16x16x32_bf16 (wave32): D = A(16x32) * B(32x16) + C
  return __builtin_amdgcn_wmma_f32_16x16x32_bf16(false, a, false, b, (short)0, c,
                                                 false, false);
}

// LDS byte offset of a __shared__ object: for the LDS aperture the low 32 bits
// of the generic address ARE the LDS address (ISA 10.2 aperture mapping).
#define LDS_OFF(p) ((uint32_t)(uintptr_t)(const void*)(p))

// ---------------------------------------------------------------------------
// Tensor Data Mover: DMA one [128 rows x 32 bf16] tile (row stride `stride0`
// elements in global) into LDS, padding 16B after each 64B row so the LDS tile
// has stride LDT = 40 bf16 (matches the WMMA fragment loader below).
// D# layout per CDNA5 ISA. Tracked by TENSORcnt.
// Builtin arity on this toolchain is 6: (g0 u32x4, g1 i32x8, g2 i32x4,
// g3 i32x4, extra i32x8, cpol i32); groups 2/3/extra unused for 2-D -> zeros.
// All inputs are workgroup-uniform so the descriptor lowers to SGPR groups.
// ---------------------------------------------------------------------------
__device__ __forceinline__ void tdm_load_tile_bf16(uint64_t ga, uint32_t lds_off,
                                                   uint32_t tdim0, uint32_t tdim1,
                                                   uint32_t stride0) {
  u32x4 g0;
  g0.x = 1u;                                  // count=1 (valid), user mode
  g0.y = lds_off;                             // lds_addr (bytes)
  g0.z = (uint32_t)ga;                        // global_addr[31:0]
  g0.w = (uint32_t)(ga >> 32) | (2u << 30);   // global_addr[56:32] | type=2
  i32x8 g1;
  g1[0] = (int)((1u << 16)        // data_size = 2 bytes
              | (1u << 20)        // pad_enable
              | (3u << 22)        // pad_interval = 16 DWORDs (one 64B tile row)
              | (3u << 25));      // pad_amount   = 4 DWORDs (16B -> 80B row stride)
  g1[1] = (int)(tdim0 << 16);                     // tensor_dim0[15:0]
  g1[2] = (int)((tdim0 >> 16) | (tdim1 << 16));   // tensor_dim0[31:16]|tensor_dim1[15:0]
  g1[3] = (int)((tdim1 >> 16) | (32u << 16));     // tensor_dim1[31:16]|tile_dim0=32
  g1[4] = 128;                                    // tile_dim1 = 128 rows (tile_dim2=0)
  g1[5] = (int)stride0;                           // tensor_dim0_stride[31:0]
  g1[6] = 0;                                      // stride0[47:32] | dim1_stride[15:0]
  g1[7] = 0;
  const i32x4 gz4 = {0, 0, 0, 0};
  const i32x8 gz8 = {0, 0, 0, 0, 0, 0, 0, 0};
  __builtin_amdgcn_tensor_load_to_lds(g0, g1, gz4, gz4, gz8, 0);
}

// ---------------------------------------------------------------- reductions
__global__ void zero_sums(float* p) {
  if (threadIdx.x < 4) p[threadIdx.x] = 0.0f;
}

__global__ __launch_bounds__(256)
void abssum_kernel(const float* __restrict__ w, float* __restrict__ out, long long n) {
  __shared__ float sh[256];
  float acc = 0.0f;
  for (long long i = (long long)blockIdx.x * 256 + threadIdx.x; i < n;
       i += (long long)gridDim.x * 256)
    acc += fabsf(w[i]);
  sh[threadIdx.x] = acc;
  __syncthreads();
  for (int s = 128; s > 0; s >>= 1) {
    if (threadIdx.x < (unsigned)s) sh[threadIdx.x] += sh[threadIdx.x + s];
    __syncthreads();
  }
  if (threadIdx.x == 0) atomicAdd(out, sh[0]);
}

// ------------------------------------------------- quantize / convert to bf16
__global__ __launch_bounds__(256)
void quant_ternary_kernel(const float* __restrict__ w, __bf16* __restrict__ t,
                          const float* __restrict__ sum, float inv_n, long long n) {
  const float s    = sum[0] * inv_n + QEPS;   // scale = mean(|W|) + eps
  const float invs = 1.0f / s;
  for (long long i = (long long)blockIdx.x * 256 + threadIdx.x; i < n;
       i += (long long)gridDim.x * 256) {
    float r = rintf(w[i] * invs);             // round-to-nearest-even (jnp.round)
    r = fminf(1.0f, fmaxf(-1.0f, r));         // clip -> ternary, exact in bf16
    t[i] = (__bf16)r;
  }
}

__global__ __launch_bounds__(256)
void cvt_bf16_kernel(const float* __restrict__ x, __bf16* __restrict__ t, long long n) {
  for (long long i = (long long)blockIdx.x * 256 + threadIdx.x; i < n;
       i += (long long)gridDim.x * 256)
    t[i] = (__bf16)x[i];
}

// --------------------------------------------------------------- GEMM 1 (fused)
// hidden[m,e] = silu(su * x.Tu^T) * (sg * x.Tg^T), bf16 out.
// Block 128x128, 8 waves (2x4), wave tile 64x32 -> 16 WMMAs/K-step, KC=32.
// TDM double-buffered LDS: wave 0 DMAs next tiles while all waves compute.
__global__ __launch_bounds__(256)
void gemm_swiglu_kernel(const __bf16* __restrict__ X,   // [M, 2048] bf16
                        const __bf16* __restrict__ WU,  // [8192, 2048] ternary bf16
                        const __bf16* __restrict__ WG,  // [8192, 2048] ternary bf16
                        __bf16* __restrict__ H,         // [M, 8192] bf16
                        const float* __restrict__ sums, int M) {
  constexpr int BM = 128, BN = 128, KC = 32, K = TOK_K, N = EXP_N, LDT = KC + 8;
  __shared__ __attribute__((aligned(16))) __bf16 xs[2][BM * LDT];
  __shared__ __attribute__((aligned(16))) __bf16 us[2][BN * LDT];
  __shared__ __attribute__((aligned(16))) __bf16 gs[2][BN * LDT];

  const int tid  = threadIdx.x;
  const int lane = tid & 31;
  const int wave = tid >> 5;
  const int wm   = wave >> 2;           // 0..1 (64 rows each)
  const int wn   = wave & 3;            // 0..3 (32 cols each)
  const int m0   = blockIdx.y * BM;
  const int n0   = blockIdx.x * BN;

  f32x8 accU[4][2] = {};
  f32x8 accG[4][2] = {};

  const int akoff = (lane & 16) ? 16 : 0;   // A layout K half
  const int bkoff = (lane & 16) ? 16 : 0;   // B layout K half

  // Uniform 64-bit base addresses (kept scalar; TDM descriptors stay in SGPRs)
  const uint64_t xga = (uint64_t)(uintptr_t)(X  + (size_t)m0 * K);
  const uint64_t uga = (uint64_t)(uintptr_t)(WU + (size_t)n0 * K);
  const uint64_t gga = (uint64_t)(uintptr_t)(WG + (size_t)n0 * K);

  if (wave == 0) {                      // prologue: DMA tile 0 into buffer 0
    tdm_load_tile_bf16(xga, LDS_OFF(&xs[0][0]), K, (uint32_t)M, K);
    tdm_load_tile_bf16(uga, LDS_OFF(&us[0][0]), K, N, K);
    tdm_load_tile_bf16(gga, LDS_OFF(&gs[0][0]), K, N, K);
  }

  for (int k0 = 0; k0 < K; k0 += KC) {
    const int cur = (k0 / KC) & 1;
    // Next-tile addresses computed in UNIFORM control flow -> scalar regs.
    const uint64_t nb  = (uint64_t)(uint32_t)((k0 + KC) * 2);   // byte offset
    const uint32_t xl  = LDS_OFF(&xs[cur ^ 1][0]);
    const uint32_t ul  = LDS_OFF(&us[cur ^ 1][0]);
    const uint32_t gl  = LDS_OFF(&gs[cur ^ 1][0]);
    const bool more    = (k0 + KC) < K;

    __syncthreads();                    // all waves done reading buf[cur^1]
    if (wave == 0) {
      if (more) {                       // DMA next tile into the freed buffer
        tdm_load_tile_bf16(xga + nb, xl, K, (uint32_t)M, K);
        tdm_load_tile_bf16(uga + nb, ul, K, N, K);
        tdm_load_tile_bf16(gga + nb, gl, K, N, K);
        __builtin_amdgcn_s_wait_tensorcnt(3);   // in-order TDM: buf[cur] complete
      } else {
        __builtin_amdgcn_s_wait_tensorcnt(0);
      }
    }
    __syncthreads();                    // buf[cur] visible to all waves

    const __bf16* xsb = &xs[cur][0];
    const __bf16* usb = &us[cur][0];
    const __bf16* gsb = &gs[cur][0];

    BF16Frag a[4], bu[2], bg[2];
#pragma unroll
    for (int mi = 0; mi < 4; ++mi) {
      const __bf16* p = xsb + (wm * 64 + mi * 16 + (lane & 15)) * LDT;
      a[mi].h[0] = *(const bf16x8*)(p + akoff);
      a[mi].h[1] = *(const bf16x8*)(p + akoff + 8);
    }
#pragma unroll
    for (int ni = 0; ni < 2; ++ni) {
      const __bf16* pu = usb + (wn * 32 + ni * 16 + (lane & 15)) * LDT + bkoff;
      bu[ni].h[0] = *(const bf16x8*)pu;
      bu[ni].h[1] = *(const bf16x8*)(pu + 8);
      const __bf16* pg = gsb + (wn * 32 + ni * 16 + (lane & 15)) * LDT + bkoff;
      bg[ni].h[0] = *(const bf16x8*)pg;
      bg[ni].h[1] = *(const bf16x8*)(pg + 8);
    }
#pragma unroll
    for (int mi = 0; mi < 4; ++mi)
#pragma unroll
      for (int ni = 0; ni < 2; ++ni) {
        accU[mi][ni] = wmma_bf16f32(a[mi].v, bu[ni].v, accU[mi][ni]);
        accG[mi][ni] = wmma_bf16f32(a[mi].v, bg[ni].v, accG[mi][ni]);
      }
  }

  const float inv_n = 1.0f / (float)((long long)EXP_N * TOK_K);
  const float su = sums[0] * inv_n + QEPS;
  const float sg = sums[1] * inv_n + QEPS;
#pragma unroll
  for (int mi = 0; mi < 4; ++mi)
#pragma unroll
    for (int ni = 0; ni < 2; ++ni) {
      const int mrow = m0 + wm * 64 + mi * 16 + ((lane & 16) ? 8 : 0);
      const int ncol = n0 + wn * 32 + ni * 16 + (lane & 15);
#pragma unroll
      for (int r = 0; r < 8; ++r) {
        float up = su * accU[mi][ni][r];
        float gt = sg * accG[mi][ni][r];
        float hv = (up / (1.0f + __expf(-up))) * gt;   // silu(up) * gate
        H[(size_t)(mrow + r) * N + ncol] = (__bf16)hv;
      }
    }
}

// --------------------------------------------------------------- GEMM 2 (down)
__global__ __launch_bounds__(256)
void gemm_down_kernel(const __bf16* __restrict__ Hd,  // [M, 8192] bf16
                      const __bf16* __restrict__ WD,  // [2048, 8192] ternary bf16
                      float* __restrict__ O,          // [M, 2048] f32
                      const float* __restrict__ sums, int M) {
  constexpr int BM = 128, BN = 128, KC = 32, K = EXP_N, N = TOK_K, LDT = KC + 8;
  __shared__ __attribute__((aligned(16))) __bf16 hs[2][BM * LDT];
  __shared__ __attribute__((aligned(16))) __bf16 ds[2][BN * LDT];

  const int tid  = threadIdx.x;
  const int lane = tid & 31;
  const int wave = tid >> 5;
  const int wm   = wave >> 2;
  const int wn   = wave & 3;
  const int m0   = blockIdx.y * BM;
  const int n0   = blockIdx.x * BN;

  f32x8 acc[4][2] = {};
  const int akoff = (lane & 16) ? 16 : 0;
  const int bkoff = (lane & 16) ? 16 : 0;

  const uint64_t hga = (uint64_t)(uintptr_t)(Hd + (size_t)m0 * K);
  const uint64_t dga = (uint64_t)(uintptr_t)(WD + (size_t)n0 * K);

  if (wave == 0) {
    tdm_load_tile_bf16(hga, LDS_OFF(&hs[0][0]), K, (uint32_t)M, K);
    tdm_load_tile_bf16(dga, LDS_OFF(&ds[0][0]), K, N, K);
  }

  for (int k0 = 0; k0 < K; k0 += KC) {
    const int cur = (k0 / KC) & 1;
    const uint64_t nb = (uint64_t)(uint32_t)((k0 + KC) * 2);
    const uint32_t hl = LDS_OFF(&hs[cur ^ 1][0]);
    const uint32_t dl = LDS_OFF(&ds[cur ^ 1][0]);
    const bool more   = (k0 + KC) < K;

    __syncthreads();
    if (wave == 0) {
      if (more) {
        tdm_load_tile_bf16(hga + nb, hl, K, (uint32_t)M, K);
        tdm_load_tile_bf16(dga + nb, dl, K, N, K);
        __builtin_amdgcn_s_wait_tensorcnt(2);
      } else {
        __builtin_amdgcn_s_wait_tensorcnt(0);
      }
    }
    __syncthreads();

    const __bf16* hsb = &hs[cur][0];
    const __bf16* dsb = &ds[cur][0];

    BF16Frag a[4], b[2];
#pragma unroll
    for (int mi = 0; mi < 4; ++mi) {
      const __bf16* p = hsb + (wm * 64 + mi * 16 + (lane & 15)) * LDT;
      a[mi].h[0] = *(const bf16x8*)(p + akoff);
      a[mi].h[1] = *(const bf16x8*)(p + akoff + 8);
    }
#pragma unroll
    for (int ni = 0; ni < 2; ++ni) {
      const __bf16* pb = dsb + (wn * 32 + ni * 16 + (lane & 15)) * LDT + bkoff;
      b[ni].h[0] = *(const bf16x8*)pb;
      b[ni].h[1] = *(const bf16x8*)(pb + 8);
    }
#pragma unroll
    for (int mi = 0; mi < 4; ++mi)
#pragma unroll
      for (int ni = 0; ni < 2; ++ni)
        acc[mi][ni] = wmma_bf16f32(a[mi].v, b[ni].v, acc[mi][ni]);
  }

  const float inv_n = 1.0f / (float)((long long)EXP_N * TOK_K);
  const float sd = sums[2] * inv_n + QEPS;
#pragma unroll
  for (int mi = 0; mi < 4; ++mi)
#pragma unroll
    for (int ni = 0; ni < 2; ++ni) {
      const int mrow = m0 + wm * 64 + mi * 16 + ((lane & 16) ? 8 : 0);
      const int ncol = n0 + wn * 32 + ni * 16 + (lane & 15);
#pragma unroll
      for (int r = 0; r < 8; ++r)
        O[(size_t)(mrow + r) * N + ncol] = sd * acc[mi][ni][r];
    }
}

// -------------------------------------------------------------------- launcher
extern "C" void kernel_launch(void* const* d_in, const int* in_sizes, int n_in,
                              void* d_out, int out_size, void* d_ws, size_t ws_size,
                              hipStream_t stream) {
  const float* x  = (const float*)d_in[0];
  const float* wu = (const float*)d_in[1];
  const float* wg = (const float*)d_in[2];
  const float* wd = (const float*)d_in[3];
  float* out = (float*)d_out;

  const int M = in_sizes[0] / TOK_K;                 // 8192 tokens
  const long long nw = (long long)EXP_N * TOK_K;     // elems per weight matrix

  char* ws = (char*)d_ws;
  float* sums = (float*)ws;                          // [0]=up [1]=gate [2]=down
  size_t off = 256;
  __bf16* xb  = (__bf16*)(ws + off); off += (size_t)M * TOK_K * 2;     // 32 MB
  __bf16* wub = (__bf16*)(ws + off); off += (size_t)nw * 2;            // 32 MB
  __bf16* wgb = (__bf16*)(ws + off); off += (size_t)nw * 2;            // 32 MB
  __bf16* wdb = (__bf16*)(ws + off); off += (size_t)nw * 2;            // 32 MB
  __bf16* hid = (__bf16*)(ws + off);                                   // 128 MB

  zero_sums<<<1, 32, 0, stream>>>(sums);
  abssum_kernel<<<1024, 256, 0, stream>>>(wu, sums + 0, nw);
  abssum_kernel<<<1024, 256, 0, stream>>>(wg, sums + 1, nw);
  abssum_kernel<<<1024, 256, 0, stream>>>(wd, sums + 2, nw);

  const float inv_n = 1.0f / (float)nw;
  quant_ternary_kernel<<<2048, 256, 0, stream>>>(wu, wub, sums + 0, inv_n, nw);
  quant_ternary_kernel<<<2048, 256, 0, stream>>>(wg, wgb, sums + 1, inv_n, nw);
  quant_ternary_kernel<<<2048, 256, 0, stream>>>(wd, wdb, sums + 2, inv_n, nw);
  cvt_bf16_kernel<<<2048, 256, 0, stream>>>(x, xb, (long long)M * TOK_K);

  gemm_swiglu_kernel<<<dim3(EXP_N / 128, M / 128), 256, 0, stream>>>(
      xb, wub, wgb, hid, sums, M);
  gemm_down_kernel<<<dim3(TOK_K / 128, M / 128), 256, 0, stream>>>(
      hid, wdb, out, sums, M);
}
